// GNN_45509473468799
// MI455X (gfx1250) — compile-verified, matrix operand order checked by reading, not run on hardware
//
#include <hip/hip_runtime.h>

#define NODES 50000
#define EDGES 800000

typedef __attribute__((ext_vector_type(16))) __bf16 v16bf;
typedef __attribute__((ext_vector_type(8)))  float  v8f;

// ---------------- device helpers ----------------

// K index held by element j (0..15) of an A/B WMMA operand for lane-group g (0/1)
// (CDNA5 ISA, 16-bit A-matrix 16x32 layout)
__device__ __forceinline__ int koff(int j, int g) {
  return (j < 8 ? j : j + 8) + (g << 3);
}

// convert 8 consecutive fp32 (16B-aligned) into bf16 operand elements [base..base+7]
__device__ __forceinline__ void cvt8(const float* __restrict__ p, v16bf& a, int base) {
  float4 x0 = *(const float4*)(p);
  float4 x1 = *(const float4*)(p + 4);
  a[base + 0] = (__bf16)x0.x; a[base + 1] = (__bf16)x0.y;
  a[base + 2] = (__bf16)x0.z; a[base + 3] = (__bf16)x0.w;
  a[base + 4] = (__bf16)x1.x; a[base + 5] = (__bf16)x1.y;
  a[base + 6] = (__bf16)x1.z; a[base + 7] = (__bf16)x1.w;
}

// stage packed B tiles (ELEMS bf16) from global into LDS, whole block cooperates.
// 32-byte chunks; ELEMS is always a multiple of 512.
template<int ELEMS>
__device__ __forceinline__ void stage_b(const __bf16* __restrict__ Bp, __bf16* Bs) {
  const int chunks = ELEMS / 16;             // 32B chunks
  for (int c = threadIdx.x; c < chunks; c += blockDim.x) {
    const uint4* s = (const uint4*)(Bp + (size_t)c * 16);
    uint4* d = (uint4*)(Bs + (size_t)c * 16);
    d[0] = s[0];
    d[1] = s[1];
  }
  __syncthreads();
}

// ---------------- weight packing ----------------
// Pack fp32 W[K][N] into WMMA B-operand tiles:
//   packed[(kt*Nt+nt)*512 + lane*16 + j] = bf16(W[kt*32 + koff(j,lane/16)][nt*16 + lane%16])
__global__ void k_pack(const float* __restrict__ W, __bf16* __restrict__ out,
                       int K, int N, int Kt, int Nt) {
  int idx = blockIdx.x * blockDim.x + threadIdx.x;
  int total = Kt * Nt * 512;
  if (idx >= total) return;
  int j    = idx & 15;
  int lane = (idx >> 4) & 31;
  int tile = idx >> 9;
  int nt = tile % Nt;
  int kt = tile / Nt;
  int g = lane >> 4;
  int n = (nt << 4) + (lane & 15);
  int k = (kt << 5) + koff(j, g);
  float v = (k < K && n < N) ? W[(size_t)k * N + n] : 0.0f;
  out[idx] = (__bf16)v;
}

// ---------------- templated GEMM: out[M,N] = act(A[M,K] @ W + bias) ----------------
// One wave computes a 16-row x (NT*16)-col tile. A fp32 row-major; W pre-packed bf16,
// staged to LDS once per block (all 8 waves share it), WMMAs fed by ds_load.
// All launches use M % 16 == 0, so no per-row guards; whole-wave early exit only
// (after the barrier), keeping EXEC all-ones for WMMA.
template<int KT, int NT, bool KFULL, bool FULLN, bool ACCUM>
__global__ __launch_bounds__(256) void k_gemm_t(
    const float* __restrict__ A, int lda, int K,
    const __bf16* __restrict__ Bp, int N,
    const float* __restrict__ bias,
    float* __restrict__ out, int ldo,
    int M, int act)
{
  __shared__ __bf16 Bs[KT * NT * 512];
  stage_b<KT * NT * 512>(Bp, Bs);

  int lane = threadIdx.x & 31;
  int wave = threadIdx.x >> 5;
  int row0 = (blockIdx.x * (blockDim.x >> 5) + wave) << 4;
  if (row0 >= M) return;
  int g = lane >> 4;
  int r = lane & 15;
  const float* Arow = A + (size_t)(row0 + r) * lda;

  v8f acc[NT] = {};
#pragma unroll
  for (int kt = 0; kt < KT; ++kt) {
    int kbase = kt << 5;
    v16bf a;
    if (KFULL) {
      const float* p = Arow + kbase + (g << 3);   // two 8-float runs, 16B aligned
      cvt8(p, a, 0);
      cvt8(p + 16, a, 8);
    } else {
#pragma unroll
      for (int j = 0; j < 16; ++j) {
        int k = kbase + koff(j, g);
        float v = (k < K) ? Arow[k] : 0.0f;
        a[j] = (__bf16)v;
      }
    }
    const __bf16* bt = Bs + ((kt * NT) << 9) + (lane << 4);
#pragma unroll
    for (int nt = 0; nt < NT; ++nt) {
      v16bf b = *(const v16bf*)(bt + (nt << 9));
      acc[nt] = __builtin_amdgcn_wmma_f32_16x16x32_bf16(
          false, a, false, b, (short)0, acc[nt], false, false);
    }
  }

  int mrow = row0 + (g << 3);
#pragma unroll
  for (int nt = 0; nt < NT; ++nt) {
    int n = (nt << 4) + r;
    if (!FULLN && n >= N) continue;
    float bv = bias ? bias[n] : 0.0f;
#pragma unroll
    for (int v = 0; v < 8; ++v) {
      float x = acc[nt][v] + bv;
      if (act) x = fmaxf(x, 0.0f);
      size_t o = (size_t)(mrow + v) * ldo + n;
      if (ACCUM) out[o] += x; else out[o] = x;
    }
  }
}

// ---------------- decoder layer-0 GEMM with fused gather-concat A ----------------
// A row e (K=192) = [ xn[src[e]] (64) | xn[dst[e]] (64) | xe[e] (64) ]
template<int NT, bool FULLN>
__global__ __launch_bounds__(256) void k_gemm_cat_t(
    const float* __restrict__ xn, const float* __restrict__ xe,
    const int* __restrict__ ei,
    const __bf16* __restrict__ Bp, int N,
    const float* __restrict__ bias,
    float* __restrict__ out, int ldo,
    int M, int act)
{
  const int KT = 6;  // K = 192
  __shared__ __bf16 Bs[KT * NT * 512];
  stage_b<KT * NT * 512>(Bp, Bs);

  int lane = threadIdx.x & 31;
  int wave = threadIdx.x >> 5;
  int row0 = (blockIdx.x * (blockDim.x >> 5) + wave) << 4;
  if (row0 >= M) return;
  int g = lane >> 4;
  int r = lane & 15;
  int e = row0 + r;
  int s = ei[2 * (size_t)e];
  int d = ei[2 * (size_t)e + 1];
  const float* s0 = xn + (size_t)s * 64;
  const float* s1 = xn + (size_t)d * 64;
  const float* s2 = xe + (size_t)e * 64;

  v8f acc[NT] = {};
#pragma unroll
  for (int kt = 0; kt < KT; ++kt) {
    int kbase = kt << 5;
    v16bf a;
    int k0 = kbase + (g << 3);                    // 8-aligned run: never crosses a segment
    const float* p0 = (k0 < 64) ? (s0 + k0) : ((k0 < 128) ? (s1 + k0 - 64) : (s2 + k0 - 128));
    int k1 = k0 + 16;
    const float* p1 = (k1 < 64) ? (s0 + k1) : ((k1 < 128) ? (s1 + k1 - 64) : (s2 + k1 - 128));
    cvt8(p0, a, 0);
    cvt8(p1, a, 8);
    const __bf16* bt = Bs + ((kt * NT) << 9) + (lane << 4);
#pragma unroll
    for (int nt = 0; nt < NT; ++nt) {
      v16bf b = *(const v16bf*)(bt + (nt << 9));
      acc[nt] = __builtin_amdgcn_wmma_f32_16x16x32_bf16(
          false, a, false, b, (short)0, acc[nt], false, false);
    }
  }

  int mrow = row0 + (g << 3);
#pragma unroll
  for (int nt = 0; nt < NT; ++nt) {
    int n = (nt << 4) + r;
    if (!FULLN && n >= N) continue;
    float bv = bias ? bias[n] : 0.0f;
#pragma unroll
    for (int v = 0; v < 8; ++v) {
      float x = acc[nt][v] + bv;
      if (act) x = fmaxf(x, 0.0f);
      out[(size_t)(mrow + v) * ldo + n] = x;
    }
  }
}

// ---------------- layernorm over 64 features, one wave per row (wave32) ----------------
__global__ void k_ln64(const float* __restrict__ x, const float* __restrict__ gm,
                       const float* __restrict__ bt, float* __restrict__ out, int M)
{
  int wid  = (int)((blockIdx.x * (size_t)blockDim.x + threadIdx.x) >> 5);
  int lane = threadIdx.x & 31;
  if (wid >= M) return;
  const float* row = x + (size_t)wid * 64;
  float a0 = row[lane], a1 = row[lane + 32];
  float s = a0 + a1, sq = a0 * a0 + a1 * a1;
#pragma unroll
  for (int off = 16; off >= 1; off >>= 1) {
    s  += __shfl_xor(s,  off, 32);
    sq += __shfl_xor(sq, off, 32);
  }
  float mu  = s * (1.0f / 64.0f);
  float var = sq * (1.0f / 64.0f) - mu * mu;
  float rs  = rsqrtf(var + 1e-5f);
  float* o = out + (size_t)wid * 64;
  o[lane]      = (a0 - mu) * rs * gm[lane]      + bt[lane];
  o[lane + 32] = (a1 - mu) * rs * gm[lane + 32] + bt[lane + 32];
}

// layernorm for small N (decoder-2, N=3): one thread per row
__global__ void k_ln_small(const float* __restrict__ x, int ldx,
                           const float* __restrict__ gm, const float* __restrict__ bt,
                           float* __restrict__ out, int ldo, int M, int N)
{
  int r = blockIdx.x * blockDim.x + threadIdx.x;
  if (r >= M) return;
  float s = 0.0f, sq = 0.0f;
  for (int i = 0; i < N; ++i) { float v = x[(size_t)r * ldx + i]; s += v; sq += v * v; }
  float mu  = s / (float)N;
  float var = sq / (float)N - mu * mu;
  float rs  = rsqrtf(var + 1e-5f);
  for (int i = 0; i < N; ++i)
    out[(size_t)r * ldo + i] = (x[(size_t)r * ldx + i] - mu) * rs * gm[i] + bt[i];
}

// ---------------- graph scatter/gather kernels ----------------
__global__ void k_zero(float* __restrict__ p, size_t n) {
  size_t i = blockIdx.x * (size_t)blockDim.x + threadIdx.x;
  if (i < n) p[i] = 0.0f;
}

__global__ void k_deg(const int* __restrict__ ei, float* __restrict__ deg, int E) {
  int e = blockIdx.x * blockDim.x + threadIdx.x;
  if (e < E) atomicAdd(&deg[ei[2 * (size_t)e]], 1.0f);
}

__global__ void k_disdiag(const float* __restrict__ deg, float* __restrict__ dis,
                          float* __restrict__ dg, int n) {
  int i = blockIdx.x * blockDim.x + threadIdx.x;
  if (i >= n) return;
  float d = deg[i];
  dis[i] = (d > 0.0f) ? rsqrtf(d) : 0.0f;
  dg[i]  = ((d > 0.0f) ? 1.0f : 0.0f) - 1.0f;
}

__global__ void k_edgew(const int* __restrict__ ei, const float* __restrict__ dis,
                        float* __restrict__ w, int E) {
  int e = blockIdx.x * blockDim.x + threadIdx.x;
  if (e >= E) return;
  w[e] = -(dis[ei[2 * (size_t)e]] * dis[ei[2 * (size_t)e + 1]]);
}

// xn[src] += xe ; xn[dst] += xe   (one thread per edge-feature)
__global__ void k_scatter_xe(const int* __restrict__ ei, const float* __restrict__ xe,
                             float* __restrict__ xn, int E) {
  size_t idx = blockIdx.x * (size_t)blockDim.x + threadIdx.x;
  if (idx >= (size_t)E * 64) return;
  int e = (int)(idx >> 6), f = (int)(idx & 63);
  float v = xe[idx];
  atomicAdd(&xn[(size_t)ei[2 * (size_t)e]     * 64 + f], v);
  atomicAdd(&xn[(size_t)ei[2 * (size_t)e + 1] * 64 + f], v);
}

// out = scale*diag*z - sub   (sub optional)
__global__ void k_prop_init(const float* __restrict__ z, const float* __restrict__ dg,
                            const float* __restrict__ sub, float scale,
                            float* __restrict__ out, int nnodes) {
  size_t idx = blockIdx.x * (size_t)blockDim.x + threadIdx.x;
  if (idx >= (size_t)nnodes * 64) return;
  int node = (int)(idx >> 6);
  float v = scale * dg[node] * z[idx];
  if (sub) v -= sub[idx];
  out[idx] = v;
}

// out[dst] += scale * w[e] * z[src]
__global__ void k_prop_edges(const int* __restrict__ ei, const float* __restrict__ w,
                             const float* __restrict__ z, float* __restrict__ out,
                             float scale, int E) {
  size_t idx = blockIdx.x * (size_t)blockDim.x + threadIdx.x;
  if (idx >= (size_t)E * 64) return;
  int e = (int)(idx >> 6), f = (int)(idx & 63);
  int s = ei[2 * (size_t)e], d = ei[2 * (size_t)e + 1];
  atomicAdd(&out[(size_t)d * 64 + f], scale * w[e] * z[(size_t)s * 64 + f]);
}

// ---------------- host orchestration ----------------
extern "C" void kernel_launch(void* const* d_in, const int* in_sizes, int n_in,
                              void* d_out, int out_size, void* d_ws, size_t ws_size,
                              hipStream_t stream)
{
  (void)in_sizes; (void)n_in; (void)out_size; (void)ws_size;
  const float* x_edge = (const float*)d_in[0];
  const int*   ei     = (const int*)d_in[1];

  // params flattened in JAX pytree order (sorted dict keys): dec < enc < gn;
  // mlp: beta, gamma, (b,w)x4 ; gn[i]: conv.bias, conv.lins, mlp
  int pidx = 2;
  auto nextp = [&]() -> const float* { return (const float*)d_in[pidx++]; };
  struct MlpP { const float* beta; const float* gamma; const float* b[4]; const float* w[4]; };
  auto readMlp = [&]() {
    MlpP m; m.beta = nextp(); m.gamma = nextp();
    for (int i = 0; i < 4; ++i) { m.b[i] = nextp(); m.w[i] = nextp(); }
    return m;
  };
  MlpP dec[3]; dec[0] = readMlp(); dec[1] = readMlp(); dec[2] = readMlp();
  MlpP enc = readMlp();
  struct GnP { const float* bias; const float* lins; MlpP mlp; };
  GnP gn[3];
  for (int i = 0; i < 3; ++i) { gn[i].bias = nextp(); gn[i].lins = nextp(); gn[i].mlp = readMlp(); }

  // ---- workspace bump allocator ----
  char* wsp = (char*)d_ws; size_t off = 0;
  auto alloc = [&](size_t bytes) -> void* {
    off = (off + 255) & ~(size_t)255;
    void* p = wsp + off; off += bytes; return p;
  };
  float* xe   = (float*)alloc((size_t)EDGES * 64 * 4);
  float* tE0  = (float*)alloc((size_t)EDGES * 64 * 4);
  float* xn   = (float*)alloc((size_t)NODES * 64 * 4);
  float* tN0  = (float*)alloc((size_t)NODES * 64 * 4);
  float* tN1  = (float*)alloc((size_t)NODES * 64 * 4);
  float* txA  = (float*)alloc((size_t)NODES * 64 * 4);
  float* txB  = (float*)alloc((size_t)NODES * 64 * 4);
  float* txC  = (float*)alloc((size_t)NODES * 64 * 4);
  float* cheb = (float*)alloc((size_t)NODES * 64 * 4);
  float* deg  = (float*)alloc((size_t)NODES * 4);
  float* dis  = (float*)alloc((size_t)NODES * 4);
  float* dgv  = (float*)alloc((size_t)NODES * 4);
  float* wbuf = (float*)alloc((size_t)EDGES * 4);

  auto pack = [&](const float* W, int K, int N) -> const __bf16* {
    int Kt = (K + 31) / 32, Nt = (N + 15) / 16;
    size_t tot = (size_t)Kt * Nt * 512;
    __bf16* p = (__bf16*)alloc(tot * 2);
    k_pack<<<dim3((unsigned)((tot + 255) / 256)), dim3(256), 0, stream>>>(W, p, K, N, Kt, Nt);
    return p;
  };

  const __bf16* encW[4];
  encW[0] = pack(enc.w[0], 13, 64);
  for (int l = 1; l < 4; ++l) encW[l] = pack(enc.w[l], 64, 64);
  const __bf16* decW[3][4];
  for (int i = 0; i < 2; ++i) {
    decW[i][0] = pack(dec[i].w[0], 192, 64);
    for (int l = 1; l < 4; ++l) decW[i][l] = pack(dec[i].w[l], 64, 64);
  }
  decW[2][0] = pack(dec[2].w[0], 192, 3);
  for (int l = 1; l < 4; ++l) decW[2][l] = pack(dec[2].w[l], 3, 3);
  const __bf16* gnW[3][4];
  const __bf16* linW[3][5];
  for (int i = 0; i < 3; ++i) {
    for (int l = 0; l < 4; ++l) gnW[i][l] = pack(gn[i].mlp.w[l], 64, 64);
    for (int k = 0; k < 5; ++k) linW[i][k] = pack(gn[i].lins + (size_t)k * 64 * 64, 64, 64);
  }

  // GEMM dispatch: shapes are static -> pick fully specialized instantiation.
  auto gemm = [&](const float* A, int lda, int K, const __bf16* Bp, int N,
                  const float* bias, float* out, int ldo, int M, int act, int accum) {
    unsigned blocks = (unsigned)((M + 127) / 128);   // 8 waves/block * 16 rows
    dim3 gb(blocks), tb(256);
    if (K == 64 && N == 64) {
      if (accum)
        k_gemm_t<2, 4, true, true, true ><<<gb, tb, 0, stream>>>(A, lda, K, Bp, N, bias, out, ldo, M, act);
      else
        k_gemm_t<2, 4, true, true, false><<<gb, tb, 0, stream>>>(A, lda, K, Bp, N, bias, out, ldo, M, act);
    } else if (K == 13) {
      k_gemm_t<1, 4, false, true, false><<<gb, tb, 0, stream>>>(A, lda, K, Bp, N, bias, out, ldo, M, act);
    } else { // K==3, N==3
      k_gemm_t<1, 1, false, false, false><<<gb, tb, 0, stream>>>(A, lda, K, Bp, N, bias, out, ldo, M, act);
    }
  };
  auto gemmcat = [&](const __bf16* Bp, int N, const float* bias,
                     float* out, int ldo, int act) {
    unsigned blocks = (unsigned)((EDGES + 127) / 128);
    if (N == 64)
      k_gemm_cat_t<4, true ><<<dim3(blocks), dim3(256), 0, stream>>>(xn, xe, ei, Bp, N, bias, out, ldo, EDGES, act);
    else
      k_gemm_cat_t<1, false><<<dim3(blocks), dim3(256), 0, stream>>>(xn, xe, ei, Bp, N, bias, out, ldo, EDGES, act);
  };
  auto ln64 = [&](const float* x, const float* g, const float* b, float* o, int M) {
    k_ln64<<<dim3((unsigned)((M + 7) / 8)), dim3(256), 0, stream>>>(x, g, b, o, M);
  };
  auto zero = [&](float* p, size_t n) {
    k_zero<<<dim3((unsigned)((n + 255) / 256)), dim3(256), 0, stream>>>(p, n);
  };

  const size_t EF = (size_t)EDGES * 64, NF = (size_t)NODES * 64;
  unsigned gEF = (unsigned)((EF + 255) / 256);
  unsigned gNF = (unsigned)((NF + 255) / 256);
  unsigned gE  = (unsigned)((EDGES + 255) / 256);
  unsigned gN  = (unsigned)((NODES + 255) / 256);

  // ---- encoder MLP: x_edge -> xe ----
  gemm(x_edge, 13, 13, encW[0], 64, enc.b[0], tE0, 64, EDGES, 1, 0);
  gemm(tE0, 64, 64, encW[1], 64, enc.b[1], xe,  64, EDGES, 1, 0);
  gemm(xe,  64, 64, encW[2], 64, enc.b[2], tE0, 64, EDGES, 1, 0);
  gemm(tE0, 64, 64, encW[3], 64, enc.b[3], xe,  64, EDGES, 0, 0);
  ln64(xe, enc.gamma, enc.beta, xe, EDGES);

  // ---- node init: xn = segsum(xe, src) + segsum(xe, dst) ----
  zero(xn, NF);
  k_scatter_xe<<<dim3(gEF), dim3(256), 0, stream>>>(ei, xe, xn, EDGES);

  // ---- degree terms (graph-static) ----
  zero(deg, NODES);
  k_deg<<<dim3(gE), dim3(256), 0, stream>>>(ei, deg, EDGES);
  k_disdiag<<<dim3(gN), dim3(256), 0, stream>>>(deg, dis, dgv, NODES);
  k_edgew<<<dim3(gE), dim3(256), 0, stream>>>(ei, dis, wbuf, EDGES);

  for (int it = 0; it < 3; ++it) {
    // ---- ChebConv: cheb = sum_k T_k(xn) @ lins[k] + bias ----
    gemm(xn, 64, 64, linW[it][0], 64, nullptr, cheb, 64, NODES, 0, 0);
    // Tx1 = prop(xn) in txA
    k_prop_init<<<dim3(gNF), dim3(256), 0, stream>>>(xn, dgv, nullptr, 1.0f, txA, NODES);
    k_prop_edges<<<dim3(gEF), dim3(256), 0, stream>>>(ei, wbuf, xn, txA, 1.0f, EDGES);
    gemm(txA, 64, 64, linW[it][1], 64, nullptr, cheb, 64, NODES, 0, 1);
    const float* T0 = xn; float* T1 = txA;
    float* ring[3] = { txB, txC, txA };
    for (int k = 2; k < 5; ++k) {
      float* T2 = ring[k - 2];
      k_prop_init<<<dim3(gNF), dim3(256), 0, stream>>>(T1, dgv, T0, 2.0f, T2, NODES);
      k_prop_edges<<<dim3(gEF), dim3(256), 0, stream>>>(ei, wbuf, T1, T2, 2.0f, EDGES);
      gemm(T2, 64, 64, linW[it][k], 64, (k == 4) ? gn[it].bias : nullptr, cheb, 64, NODES, 0, 1);
      T0 = T1; T1 = T2;
    }
    // ---- node MLP: cheb -> xn ----
    gemm(cheb, 64, 64, gnW[it][0], 64, gn[it].mlp.b[0], tN0, 64, NODES, 1, 0);
    gemm(tN0,  64, 64, gnW[it][1], 64, gn[it].mlp.b[1], tN1, 64, NODES, 1, 0);
    gemm(tN1,  64, 64, gnW[it][2], 64, gn[it].mlp.b[2], tN0, 64, NODES, 1, 0);
    gemm(tN0,  64, 64, gnW[it][3], 64, gn[it].mlp.b[3], tN1, 64, NODES, 0, 0);
    ln64(tN1, gn[it].mlp.gamma, gn[it].mlp.beta, xn, NODES);

    // ---- edge decoder: xe = MLP([xn[src], xn[dst], xe]) ----
    if (it < 2) {
      gemmcat(decW[it][0], 64, dec[it].b[0], tE0, 64, 1);       // consumes old xe
      gemm(tE0, 64, 64, decW[it][1], 64, dec[it].b[1], xe,  64, EDGES, 1, 0);
      gemm(xe,  64, 64, decW[it][2], 64, dec[it].b[2], tE0, 64, EDGES, 1, 0);
      gemm(tE0, 64, 64, decW[it][3], 64, dec[it].b[3], xe,  64, EDGES, 0, 0);
      ln64(xe, dec[it].gamma, dec[it].beta, xe, EDGES);
    } else {
      gemmcat(decW[2][0], 3, dec[2].b[0], tE0, 16, 1);          // N=3 padded to one 16-col tile
      gemm(tE0, 16, 3, decW[2][1], 3, dec[2].b[1], xe,  16, EDGES, 1, 0);
      gemm(xe,  16, 3, decW[2][2], 3, dec[2].b[2], tE0, 16, EDGES, 1, 0);
      gemm(tE0, 16, 3, decW[2][3], 3, dec[2].b[3], xe,  16, EDGES, 0, 0);
      k_ln_small<<<dim3(gE), dim3(256), 0, stream>>>(xe, 16, dec[2].gamma, dec[2].beta,
                                                     (float*)d_out, 3, EDGES, 3);
    }
  }
}
